// GraphNetBlock_63445256896966
// MI455X (gfx1250) — matrix operand order repaired and checked by reading, not compiled, not run on hardware
//
#include <hip/hip_runtime.h>

// ---------------------------------------------------------------------------
// GraphNetBlock for MI455X (gfx1250): bf16 WMMA (v_wmma_f32_16x16x32_bf16)
// - Each wave owns TWO 16-row M-tiles (32 rows): every B fragment feeds 2 WMMAs.
// - B fragments are software double-buffered so their load latency overlaps
//   with the current WMMA pair instead of stalling at s_wait_loadcnt 0.
// ---------------------------------------------------------------------------

typedef __attribute__((ext_vector_type(16))) __bf16 bf16x16;
typedef __attribute__((ext_vector_type(8)))  __bf16 bf16x8;
typedef __attribute__((ext_vector_type(8)))  float  f32x8;

#define DIM     128
#define LN_EPS  1e-5f

// Convert 8 consecutive f32 from p into a[base..base+7] as bf16.
__device__ __forceinline__ void load8_cvt(bf16x16& a, int base, const float* __restrict__ p) {
    float4 x = *reinterpret_cast<const float4*>(p);
    float4 y = *reinterpret_cast<const float4*>(p + 4);
    a[base + 0] = (__bf16)x.x; a[base + 1] = (__bf16)x.y;
    a[base + 2] = (__bf16)x.z; a[base + 3] = (__bf16)x.w;
    a[base + 4] = (__bf16)y.x; a[base + 5] = (__bf16)y.y;
    a[base + 6] = (__bf16)y.z; a[base + 7] = (__bf16)y.w;
}

// Load one packed B fragment (32 lanes x 16 bf16, contiguous 32B per lane).
__device__ __forceinline__ bf16x16 loadB(const __bf16* __restrict__ p, int f, int lane) {
    return *reinterpret_cast<const bf16x16*>(p + ((size_t)f * 32 + lane) * 16);
}

// ---------------------------------------------------------------------------
// Prep: zero the aggregation buffer (segment-sum target).
// ---------------------------------------------------------------------------
__global__ void zero_kernel(float4* __restrict__ p, long n4) {
    long i = (long)blockIdx.x * blockDim.x + threadIdx.x;
    if (i < n4) p[i] = make_float4(0.f, 0.f, 0.f, 0.f);
}

// ---------------------------------------------------------------------------
// Prep: pack a [K x 128] f32 weight matrix into bf16 WMMA B-fragment order.
// Fragment f = kt*8 + nt. Lane l: n = nt*16 + (l&15), khalf = l>>4.
// Element j (0..15): K = kt*32 + 8*khalf + j + (j >= 8 ? 8 : 0)   (dense bf16 B layout, wave32).
// ---------------------------------------------------------------------------
__global__ void pack_w_kernel(const float* __restrict__ w, __bf16* __restrict__ out,
                              int ktiles) {
    int t = blockIdx.x * blockDim.x + threadIdx.x;
    int nthreads = ktiles * 8 * 32;
    if (t >= nthreads) return;
    int f    = t >> 5;
    int lane = t & 31;
    int kt   = f >> 3;
    int nt   = f & 7;
    int n    = nt * 16 + (lane & 15);
    int k0   = kt * 32 + 8 * (lane >> 4);
    __bf16* o = out + (size_t)t * 16;
#pragma unroll
    for (int j = 0; j < 16; ++j)
        o[j] = (__bf16)w[(size_t)(k0 + j + (j >= 8 ? 8 : 0)) * DIM + n];
}

// ---------------------------------------------------------------------------
// Edge update: [src|dst|edge] (384) -> 128 ReLU -> 128 -> LayerNorm.
// Writes residual output and atomically accumulates new_edge into agg[recv].
// One wave = 32 edges (2 M-tiles); block = 4 waves = 128 edges.
// ---------------------------------------------------------------------------
__global__ __launch_bounds__(128) void edge_kernel(
    const float* __restrict__ nodef, const float* __restrict__ edgef,
    const int*  __restrict__ senders, const int* __restrict__ receivers,
    const __bf16* __restrict__ w1p, const float* __restrict__ b1,
    const __bf16* __restrict__ w2p, const float* __restrict__ b2,
    const float* __restrict__ gvec, const float* __restrict__ betav,
    float* __restrict__ out_edge, float* __restrict__ agg, long E)
{
    __shared__ __bf16 hbuf[4][32][136];   // padded rows: 272B stride, 16B aligned

    const int wave = threadIdx.x >> 5;
    const int lane = threadIdx.x & 31;
    const int ln   = lane & 15;
    const int hi   = lane >> 4;
    const long e0  = (long)blockIdx.x * 128 + wave * 32;
    if (e0 >= E) return;

    // Per-lane gather pointers for both M-tiles.
    const float* srcp[2];
    const float* dstp[2];
    const float* edp[2];
#pragma unroll
    for (int t = 0; t < 2; ++t) {
        long em = e0 + t * 16 + ln;
        srcp[t] = nodef + (long)senders[em]   * DIM;
        dstp[t] = nodef + (long)receivers[em] * DIM;
        edp[t]  = edgef + em * DIM;
    }

    // ---------------- GEMM 1: [32x384] x [384x128] ----------------
    f32x8 acc[2][8];
#pragma unroll
    for (int t = 0; t < 2; ++t)
#pragma unroll
        for (int i = 0; i < 8; ++i) acc[t][i] = f32x8{};

    bf16x16 bcur = loadB(w1p, 0, lane);            // prime the B pipeline
#pragma unroll
    for (int kt = 0; kt < 12; ++kt) {
        int K0 = kt * 32 + 8 * hi;        // A layout: elems 0..7 -> K0..K0+7
        int K1 = K0 + 16;                 //           elems 8..15 -> K1..K1+7
        bf16x16 a[2];
#pragma unroll
        for (int t = 0; t < 2; ++t) {
            const float* p0 = (K0 < 128) ? srcp[t] + K0
                             : (K0 < 256) ? dstp[t] + (K0 - 128) : edp[t] + (K0 - 256);
            const float* p1 = (K1 < 128) ? srcp[t] + K1
                             : (K1 < 256) ? dstp[t] + (K1 - 128) : edp[t] + (K1 - 256);
            load8_cvt(a[t], 0, p0);
            load8_cvt(a[t], 8, p1);
        }
#pragma unroll
        for (int nt = 0; nt < 8; ++nt) {
            int fi = kt * 8 + nt;
            bf16x16 bnext = (fi + 1 < 96) ? loadB(w1p, fi + 1, lane) : bcur;
            acc[0][nt] = __builtin_amdgcn_wmma_f32_16x16x32_bf16(
                false, a[0], false, bcur, (short)0, acc[0][nt], false, false);
            acc[1][nt] = __builtin_amdgcn_wmma_f32_16x16x32_bf16(
                false, a[1], false, bcur, (short)0, acc[1][nt], false, false);
            bcur = bnext;
        }
    }

    // bias + ReLU, stage bf16 h tiles in LDS (C-layout -> row-major re-layout)
#pragma unroll
    for (int nt = 0; nt < 8; ++nt) {
        float bias = b1[nt * 16 + ln];
#pragma unroll
        for (int t = 0; t < 2; ++t)
#pragma unroll
            for (int r = 0; r < 8; ++r) {
                float v = fmaxf(acc[t][nt][r] + bias, 0.0f);
                hbuf[wave][t * 16 + r + 8 * hi][nt * 16 + ln] = (__bf16)v;
            }
    }
    __builtin_amdgcn_wave_barrier();       // same-wave DS ops are in-order

    // ---------------- GEMM 2: [32x128] x [128x128] ----------------
    f32x8 acc2[2][8];
#pragma unroll
    for (int t = 0; t < 2; ++t)
#pragma unroll
        for (int i = 0; i < 8; ++i) acc2[t][i] = f32x8{};

    const __bf16* hrow0 = &hbuf[wave][ln][0];
    const __bf16* hrow1 = &hbuf[wave][16 + ln][0];
    bcur = loadB(w2p, 0, lane);
#pragma unroll
    for (int kt = 0; kt < 4; ++kt) {
        bf16x16 a[2];
#pragma unroll
        for (int t = 0; t < 2; ++t) {
            const __bf16* hr = t ? hrow1 : hrow0;
            bf16x8 g0 = *reinterpret_cast<const bf16x8*>(hr + kt * 32 + 8 * hi);
            bf16x8 g1 = *reinterpret_cast<const bf16x8*>(hr + kt * 32 + 16 + 8 * hi);
#pragma unroll
            for (int j = 0; j < 8; ++j) { a[t][j] = g0[j]; a[t][j + 8] = g1[j]; }
        }
#pragma unroll
        for (int nt = 0; nt < 8; ++nt) {
            int fi = kt * 8 + nt;
            bf16x16 bnext = (fi + 1 < 32) ? loadB(w2p, fi + 1, lane) : bcur;
            acc2[0][nt] = __builtin_amdgcn_wmma_f32_16x16x32_bf16(
                false, a[0], false, bcur, (short)0, acc2[0][nt], false, false);
            acc2[1][nt] = __builtin_amdgcn_wmma_f32_16x16x32_bf16(
                false, a[1], false, bcur, (short)0, acc2[1][nt], false, false);
            bcur = bnext;
        }
    }

    // ---------------- bias + LayerNorm + residual + scatter-add ----------------
    float b2v[8], gv[8], bv[8];
#pragma unroll
    for (int nt = 0; nt < 8; ++nt) {
        int n = nt * 16 + ln;
        b2v[nt] = b2[n]; gv[nt] = gvec[n]; bv[nt] = betav[n];
    }

#pragma unroll
    for (int t = 0; t < 2; ++t) {
#pragma unroll
        for (int r = 0; r < 8; ++r) {      // row m = t*16 + r + 8*hi of the tile
            float vals[8], s1 = 0.f, s2 = 0.f;
#pragma unroll
            for (int nt = 0; nt < 8; ++nt) {
                float v = acc2[t][nt][r] + b2v[nt];
                vals[nt] = v; s1 += v; s2 += v * v;
            }
#pragma unroll
            for (int mk = 1; mk < 16; mk <<= 1) {   // reduce across 16-lane half
                s1 += __shfl_xor(s1, mk);
                s2 += __shfl_xor(s2, mk);
            }
            float mean = s1 * (1.0f / DIM);
            float varr = fmaxf(s2 * (1.0f / DIM) - mean * mean, 0.0f);
            float inv  = rsqrtf(varr + LN_EPS);

            long er  = e0 + t * 16 + r + 8 * hi;
            int  rcv = receivers[er];
            const float* ef = edgef    + er * DIM;
            float*       op = out_edge + er * DIM;
            float*       ap = agg + (long)rcv * DIM;
#pragma unroll
            for (int nt = 0; nt < 8; ++nt) {
                int n = nt * 16 + ln;
                float ne = (vals[nt] - mean) * inv * gv[nt] + bv[nt];
                op[n] = ne + ef[n];        // residual edge output (coalesced)
                atomicAdd(ap + n, ne);     // segment-sum (coalesced per row)
            }
        }
    }
}

// ---------------------------------------------------------------------------
// Node update: [node|agg] (256) -> 128 ReLU -> 128 -> LayerNorm -> +residual.
// One wave = 32 nodes (2 M-tiles); block = 128 nodes.
// ---------------------------------------------------------------------------
__global__ __launch_bounds__(128) void node_kernel(
    const float* __restrict__ nodef, const float* __restrict__ agg,
    const __bf16* __restrict__ w1p, const float* __restrict__ b1,
    const __bf16* __restrict__ w2p, const float* __restrict__ b2,
    const float* __restrict__ gvec, const float* __restrict__ betav,
    float* __restrict__ out_node, long N)
{
    __shared__ __bf16 hbuf[4][32][136];

    const int wave = threadIdx.x >> 5;
    const int lane = threadIdx.x & 31;
    const int ln   = lane & 15;
    const int hi   = lane >> 4;
    const long v0  = (long)blockIdx.x * 128 + wave * 32;
    if (v0 >= N) return;                   // whole-wave uniform skip

    const float* nfp[2];
    const float* agp[2];
#pragma unroll
    for (int t = 0; t < 2; ++t) {
        long vm = v0 + t * 16 + ln;
        nfp[t] = nodef + vm * DIM;
        agp[t] = agg   + vm * DIM;
    }

    f32x8 acc[2][8];
#pragma unroll
    for (int t = 0; t < 2; ++t)
#pragma unroll
        for (int i = 0; i < 8; ++i) acc[t][i] = f32x8{};

    bf16x16 bcur = loadB(w1p, 0, lane);
#pragma unroll
    for (int kt = 0; kt < 8; ++kt) {       // K = 256
        int K0 = kt * 32 + 8 * hi;
        int K1 = K0 + 16;
        bf16x16 a[2];
#pragma unroll
        for (int t = 0; t < 2; ++t) {
            const float* p0 = (K0 < 128) ? nfp[t] + K0 : agp[t] + (K0 - 128);
            const float* p1 = (K1 < 128) ? nfp[t] + K1 : agp[t] + (K1 - 128);
            load8_cvt(a[t], 0, p0);
            load8_cvt(a[t], 8, p1);
        }
#pragma unroll
        for (int nt = 0; nt < 8; ++nt) {
            int fi = kt * 8 + nt;
            bf16x16 bnext = (fi + 1 < 64) ? loadB(w1p, fi + 1, lane) : bcur;
            acc[0][nt] = __builtin_amdgcn_wmma_f32_16x16x32_bf16(
                false, a[0], false, bcur, (short)0, acc[0][nt], false, false);
            acc[1][nt] = __builtin_amdgcn_wmma_f32_16x16x32_bf16(
                false, a[1], false, bcur, (short)0, acc[1][nt], false, false);
            bcur = bnext;
        }
    }

#pragma unroll
    for (int nt = 0; nt < 8; ++nt) {
        float bias = b1[nt * 16 + ln];
#pragma unroll
        for (int t = 0; t < 2; ++t)
#pragma unroll
            for (int r = 0; r < 8; ++r) {
                float v = fmaxf(acc[t][nt][r] + bias, 0.0f);
                hbuf[wave][t * 16 + r + 8 * hi][nt * 16 + ln] = (__bf16)v;
            }
    }
    __builtin_amdgcn_wave_barrier();

    f32x8 acc2[2][8];
#pragma unroll
    for (int t = 0; t < 2; ++t)
#pragma unroll
        for (int i = 0; i < 8; ++i) acc2[t][i] = f32x8{};

    const __bf16* hrow0 = &hbuf[wave][ln][0];
    const __bf16* hrow1 = &hbuf[wave][16 + ln][0];
    bcur = loadB(w2p, 0, lane);
#pragma unroll
    for (int kt = 0; kt < 4; ++kt) {
        bf16x16 a[2];
#pragma unroll
        for (int t = 0; t < 2; ++t) {
            const __bf16* hr = t ? hrow1 : hrow0;
            bf16x8 g0 = *reinterpret_cast<const bf16x8*>(hr + kt * 32 + 8 * hi);
            bf16x8 g1 = *reinterpret_cast<const bf16x8*>(hr + kt * 32 + 16 + 8 * hi);
#pragma unroll
            for (int j = 0; j < 8; ++j) { a[t][j] = g0[j]; a[t][j + 8] = g1[j]; }
        }
#pragma unroll
        for (int nt = 0; nt < 8; ++nt) {
            int fi = kt * 8 + nt;
            bf16x16 bnext = (fi + 1 < 32) ? loadB(w2p, fi + 1, lane) : bcur;
            acc2[0][nt] = __builtin_amdgcn_wmma_f32_16x16x32_bf16(
                false, a[0], false, bcur, (short)0, acc2[0][nt], false, false);
            acc2[1][nt] = __builtin_amdgcn_wmma_f32_16x16x32_bf16(
                false, a[1], false, bcur, (short)0, acc2[1][nt], false, false);
            bcur = bnext;
        }
    }

    float b2v[8], gv[8], bv[8];
#pragma unroll
    for (int nt = 0; nt < 8; ++nt) {
        int n = nt * 16 + ln;
        b2v[nt] = b2[n]; gv[nt] = gvec[n]; bv[nt] = betav[n];
    }

#pragma unroll
    for (int t = 0; t < 2; ++t) {
#pragma unroll
        for (int r = 0; r < 8; ++r) {
            float vals[8], s1 = 0.f, s2 = 0.f;
#pragma unroll
            for (int nt = 0; nt < 8; ++nt) {
                float v = acc2[t][nt][r] + b2v[nt];
                vals[nt] = v; s1 += v; s2 += v * v;
            }
#pragma unroll
            for (int mk = 1; mk < 16; mk <<= 1) {
                s1 += __shfl_xor(s1, mk);
                s2 += __shfl_xor(s2, mk);
            }
            float mean = s1 * (1.0f / DIM);
            float varr = fmaxf(s2 * (1.0f / DIM) - mean * mean, 0.0f);
            float inv  = rsqrtf(varr + LN_EPS);

            long vr = v0 + t * 16 + r + 8 * hi;
            const float* nf = nodef    + vr * DIM;
            float*       op = out_node + vr * DIM;
#pragma unroll
            for (int nt = 0; nt < 8; ++nt) {
                int n = nt * 16 + ln;
                float nn = (vals[nt] - mean) * inv * gv[nt] + bv[nt];
                op[n] = nn + nf[n];        // residual node output
            }
        }
    }
}

// ---------------------------------------------------------------------------
// Launch
// ---------------------------------------------------------------------------
extern "C" void kernel_launch(void* const* d_in, const int* in_sizes, int n_in,
                              void* d_out, int out_size, void* d_ws, size_t ws_size,
                              hipStream_t stream) {
    const float* nodef   = (const float*)d_in[0];
    const float* edgef   = (const float*)d_in[1];
    const int*   senders = (const int*)  d_in[2];
    const int*   recvrs  = (const int*)  d_in[3];
    const float* e_w1 = (const float*)d_in[4];
    const float* e_b1 = (const float*)d_in[5];
    const float* e_w2 = (const float*)d_in[6];
    const float* e_b2 = (const float*)d_in[7];
    const float* e_g  = (const float*)d_in[8];
    const float* e_be = (const float*)d_in[9];
    const float* n_w1 = (const float*)d_in[10];
    const float* n_b1 = (const float*)d_in[11];
    const float* n_w2 = (const float*)d_in[12];
    const float* n_b2 = (const float*)d_in[13];
    const float* n_g  = (const float*)d_in[14];
    const float* n_be = (const float*)d_in[15];

    const long N = in_sizes[0] / DIM;
    const long E = in_sizes[1] / DIM;

    float* out_node = (float*)d_out;
    float* out_edge = out_node + (size_t)N * DIM;

    // Workspace layout: agg[N*128 f32] | packed bf16 weights
    char* ws = (char*)d_ws;
    float* agg = (float*)ws;
    size_t off = (size_t)N * DIM * sizeof(float);
    __bf16* ew1p = (__bf16*)(ws + off); off += (size_t)3 * DIM * DIM * sizeof(__bf16);
    __bf16* ew2p = (__bf16*)(ws + off); off += (size_t)DIM * DIM * sizeof(__bf16);
    __bf16* nw1p = (__bf16*)(ws + off); off += (size_t)2 * DIM * DIM * sizeof(__bf16);
    __bf16* nw2p = (__bf16*)(ws + off);

    // 1) zero segment-sum buffer (every call: determinism)
    long n4 = (long)N * DIM / 4;
    zero_kernel<<<(int)((n4 + 255) / 256), 256, 0, stream>>>((float4*)agg, n4);

    // 2) pack weights into WMMA B-fragment layout (bf16)
    pack_w_kernel<<<(12 * 8 * 32 + 127) / 128, 128, 0, stream>>>(e_w1, ew1p, 12);
    pack_w_kernel<<<( 4 * 8 * 32 + 127) / 128, 128, 0, stream>>>(e_w2, ew2p, 4);
    pack_w_kernel<<<( 8 * 8 * 32 + 127) / 128, 128, 0, stream>>>(n_w1, nw1p, 8);
    pack_w_kernel<<<( 4 * 8 * 32 + 127) / 128, 128, 0, stream>>>(n_w2, nw2p, 4);

    // 3) edge MLP + residual + scatter-add into agg
    edge_kernel<<<(int)((E + 127) / 128), 128, 0, stream>>>(
        nodef, edgef, senders, recvrs,
        ew1p, e_b1, ew2p, e_b2, e_g, e_be,
        out_edge, agg, E);

    // 4) node MLP + residual
    node_kernel<<<(int)((N + 127) / 128), 128, 0, stream>>>(
        nodef, agg,
        nw1p, n_b1, nw2p, n_b2, n_g, n_be,
        out_node, N);
}
